// MoE_41540923687569
// MI455X (gfx1250) — matrix operand order repaired and checked by reading, not compiled, not run on hardware
//
#include <hip/hip_runtime.h>
#include <hip/hip_bf16.h>

#define E     8
#define DIN   1024
#define DHID  4096
#define DOUT  1024
#define NTOK  16384
#define CHUNK 1024            // hidden-dim chunk (4 chunks of 1024)

typedef __attribute__((ext_vector_type(16))) __bf16 v16bf;
typedef __attribute__((ext_vector_type(8)))  float  v8f;

static __device__ __forceinline__ __bf16 f2bf(float f) {
    union { float f; unsigned u; } v;
    v.f = f;
    unsigned r = (v.u + 0x7FFFu + ((v.u >> 16) & 1u)) >> 16;
    unsigned short s = (unsigned short)r;
    return __builtin_bit_cast(__bf16, s);
}

// ---------------- zero out + counters ----------------
__global__ void moe_zero(float* __restrict__ out, int n, int* __restrict__ counts) {
    int i = blockIdx.x * 256 + threadIdx.x;
    if (i < E) counts[i] = 0;
    for (; i < n; i += gridDim.x * 256) out[i] = 0.f;
}

// ---------------- fp32 -> bf16 weight swizzle into WMMA B-fragment layout ----------------
// dst element index = (((e*Ktiles + kt)*Ntiles + nt)*32 + lane)*16 + j
// maps to src[e][k][n] with k = kt*32 + (lane<16?0:8) + (j&7) + (j>=8?16:0), n = nt*16 + (lane&15)
__global__ void moe_swz(const float* __restrict__ src, __bf16* __restrict__ dst, int K, int N) {
    const long total = (long)E * K * N;
    const int ntiles = N >> 4;
    const int ktiles = K >> 5;
    for (long idx = (long)blockIdx.x * 256 + threadIdx.x; idx < total;
         idx += (long)gridDim.x * 256) {
        int  j    = (int)(idx & 15);
        int  lane = (int)((idx >> 4) & 31);
        long rest = idx >> 9;
        int  nt   = (int)(rest % ntiles); rest /= ntiles;
        int  kt   = (int)(rest % ktiles);
        int  e    = (int)(rest / ktiles);
        int  k    = kt * 32 + ((lane < 16) ? 0 : 8) + (j & 7) + ((j >= 8) ? 16 : 0);
        int  n    = nt * 16 + (lane & 15);
        dst[idx] = f2bf(src[(size_t)(e * K + k) * N + n]);
    }
}

// ---------------- gating: one wave32 per token ----------------
__global__ __launch_bounds__(256) void moe_gate(const float* __restrict__ x,
                                                const float* __restrict__ gw,
                                                const float* __restrict__ gb,
                                                int* __restrict__ counts,
                                                int* __restrict__ ids,
                                                float* __restrict__ tws) {
    const int t    = blockIdx.x * 8 + (threadIdx.x >> 5);
    const int lane = threadIdx.x & 31;
    if (t >= NTOK) return;

    float acc[E] = {};
    const float* row = x + (size_t)t * DIN;
    for (int i = lane; i < DIN; i += 32) {
        float xv = row[i];
        const float* g = gw + i * E;
#pragma unroll
        for (int e = 0; e < E; ++e) acc[e] += xv * g[e];
    }
#pragma unroll
    for (int off = 16; off > 0; off >>= 1) {
#pragma unroll
        for (int e = 0; e < E; ++e) acc[e] += __shfl_xor(acc[e], off, 32);
    }
    if (lane == 0) {
        float best = -1e30f, sec = -1e30f;
        int bi = 0, si = 0;
#pragma unroll
        for (int e = 0; e < E; ++e) {
            float v = acc[e] + gb[e];
            if (v > best)      { sec = best; si = bi; best = v; bi = e; }
            else if (v > sec)  { sec = v;  si = e; }
        }
        float ex = __expf(sec - best);
        float inv = 1.f / (1.f + ex);
        int s0 = atomicAdd(counts + bi, 1);
        ids[bi * NTOK + s0] = t; tws[bi * NTOK + s0] = inv;
        int s1 = atomicAdd(counts + si, 1);
        ids[si * NTOK + s1] = t; tws[si * NTOK + s1] = ex * inv;
    }
}

// ---------------- fused FFN: 32 tokens/block (2 M-tiles), 16 waves, bf16 WMMA ----------------
// Each B fragment load feeds TWO wmma ops (two M-tiles) -> halved L2 B-traffic.
// LDS: xFrag 64KB | hFrag (1 hidden chunk) 64KB | yAcc fp32 128KB | ids/weights
#define MT 2                                     // M-tiles per block (32 tokens)
#define FRAG_ELEMS (32 * 16)                     // bf16 elems per fragment (32 lanes x 16)
#define XFRAG_ELEMS (MT * (DIN / 32) * FRAG_ELEMS)
#define HFRAG_ELEMS (MT * (CHUNK / 32) * FRAG_ELEMS)
#define LDS_XFRAG_BYTES (XFRAG_ELEMS * 2)
#define LDS_HFRAG_BYTES (HFRAG_ELEMS * 2)
#define LDS_YACC_BYTES  (MT * 16 * DOUT * 4)

__global__ __launch_bounds__(512) void moe_ffn(const float* __restrict__ x,
                                               const float* __restrict__ b1,
                                               const float* __restrict__ b2,
                                               const v16bf* __restrict__ w1s,
                                               const v16bf* __restrict__ w2s,
                                               const int* __restrict__ counts,
                                               const int* __restrict__ ids,
                                               const float* __restrict__ tws,
                                               float* __restrict__ out) {
    const int e    = blockIdx.y;
    const int cnt  = counts[e];
    const int row0 = blockIdx.x * (MT * 16);
    if (row0 >= cnt) return;

    extern __shared__ __attribute__((aligned(32))) char smem[];
    __bf16* xFrag = (__bf16*)smem;
    __bf16* hFrag = (__bf16*)(smem + LDS_XFRAG_BYTES);
    float*  yAcc  = (float*)(smem + LDS_XFRAG_BYTES + LDS_HFRAG_BYTES);
    int*    sTok  = (int*)(smem + LDS_XFRAG_BYTES + LDS_HFRAG_BYTES + LDS_YACC_BYTES);
    float*  sW    = (float*)(sTok + MT * 16);

    const int tid  = threadIdx.x;
    const int lane = tid & 31;
    const int wave = tid >> 5;           // 0..15
    const int n_lo = lane & 15;
    const int mB   = (lane < 16) ? 0 : 8;

    if (tid < MT * 16) {
        int g = row0 + tid;
        if (g < cnt) { sTok[tid] = ids[e * NTOK + g]; sW[tid] = tws[e * NTOK + g]; }
        else         { sTok[tid] = 0;                 sW[tid] = 0.f; }   // pad row, weight 0
    }
    __syncthreads();

    // Gather x (fp32 -> bf16) into A-fragment layout, and zero the y accumulator slab.
    for (int p = tid; p < MT * (DIN / 32) * 32; p += 512) {
        int mtile = p >> 10;             // (DIN/32)*32 = 1024 pairs per mtile
        int rest  = p & 1023;
        int kt    = rest >> 5;
        int l     = rest & 31;
        int m     = mtile * 16 + (l & 15);
        int kb    = (l < 16) ? 0 : 8;
        const float* row = x + (size_t)sTok[m] * DIN + kt * 32 + kb;
        __bf16* dst = xFrag + p * 16;
#pragma unroll
        for (int j = 0; j < 8; ++j) dst[j]     = f2bf(row[j]);
#pragma unroll
        for (int j = 0; j < 8; ++j) dst[8 + j] = f2bf(row[16 + j]);
    }
    for (int p = tid; p < MT * 16 * DOUT; p += 512) yAcc[p] = 0.f;
    __syncthreads();

    // ---- Hidden-dimension chunk loop ----
    for (int c = 0; c < DHID / CHUNK; ++c) {
        // Phase A: h_chunk[32, CHUNK] = relu(x @ W1[:, chunk] + b1), -> LDS A-frag layout
        for (int ntl = wave; ntl < CHUNK / 16; ntl += 16) {
            const int nt = c * (CHUNK / 16) + ntl;   // global n-tile in DHID
            v8f acc0 = {}, acc1 = {};
            for (int kt = 0; kt < DIN / 32; ++kt) {
                v16bf b  = w1s[((size_t)((e * (DIN / 32) + kt) * (DHID / 16) + nt)) * 32 + lane];
                v16bf a0 = *(const v16bf*)(xFrag + (kt * 32 + lane) * 16);
                v16bf a1 = *(const v16bf*)(xFrag + ((DIN / 32) * 32 + kt * 32 + lane) * 16);
                acc0 = __builtin_amdgcn_wmma_f32_16x16x32_bf16(false, a0, false, b,
                                                               (short)0, acc0, false, false);
                acc1 = __builtin_amdgcn_wmma_f32_16x16x32_bf16(false, a1, false, b,
                                                               (short)0, acc1, false, false);
            }
            const int   hl   = ntl * 16 + n_lo;           // column within chunk
            const float bias = b1[e * DHID + c * CHUNK + hl];
            const int kt2 = hl >> 5;
            const int kk  = hl & 31;
            const int j2  = (kk & 7) + ((kk >= 16) ? 8 : 0);
            const int lb  = (((kk & 15) >= 8) ? 16 : 0);
#pragma unroll
            for (int j = 0; j < 8; ++j) {
                float v0 = acc0[j] + bias; v0 = v0 > 0.f ? v0 : 0.f;
                float v1 = acc1[j] + bias; v1 = v1 > 0.f ? v1 : 0.f;
                hFrag[(kt2 * 32 + lb + mB + j) * 16 + j2] = f2bf(v0);
                hFrag[((CHUNK / 32) * 32 + kt2 * 32 + lb + mB + j) * 16 + j2] = f2bf(v1);
            }
        }
        __syncthreads();

        // Phase B partial: yAcc += h_chunk @ W2[chunk, :]
        for (int ntB = wave; ntB < DOUT / 16; ntB += 16) {
            v8f acc0 = {}, acc1 = {};
            for (int kt2 = 0; kt2 < CHUNK / 32; ++kt2) {
                const int ktg = c * (CHUNK / 32) + kt2;   // global k-tile in DHID
                v16bf b  = w2s[((size_t)((e * (DHID / 32) + ktg) * (DOUT / 16) + ntB)) * 32 + lane];
                v16bf a0 = *(const v16bf*)(hFrag + (kt2 * 32 + lane) * 16);
                v16bf a1 = *(const v16bf*)(hFrag + ((CHUNK / 32) * 32 + kt2 * 32 + lane) * 16);
                acc0 = __builtin_amdgcn_wmma_f32_16x16x32_bf16(false, a0, false, b,
                                                               (short)0, acc0, false, false);
                acc1 = __builtin_amdgcn_wmma_f32_16x16x32_bf16(false, a1, false, b,
                                                               (short)0, acc1, false, false);
            }
            const int col = ntB * 16 + n_lo;
#pragma unroll
            for (int j = 0; j < 8; ++j) {
                yAcc[(mB + j) * DOUT + col]        += acc0[j];   // rows  0..15 (wave-owned cols)
                yAcc[(16 + mB + j) * DOUT + col]   += acc1[j];   // rows 16..31
            }
        }
        __syncthreads();   // protect hFrag reuse next chunk; yAcc cols are wave-private
    }

    // ---- Epilogue: bias, gate-weight scale, atomic combine into out ----
    for (int p = tid; p < MT * 16 * DOUT; p += 512) {
        int row = p >> 10;                 // DOUT = 1024
        int col = p & (DOUT - 1);
        float v = (yAcc[p] + b2[e * DOUT + col]) * sW[row];
        atomicAdd(out + (size_t)sTok[row] * DOUT + col, v);
    }
}

extern "C" void kernel_launch(void* const* d_in, const int* in_sizes, int n_in,
                              void* d_out, int out_size, void* d_ws, size_t ws_size,
                              hipStream_t stream) {
    (void)in_sizes; (void)n_in; (void)out_size; (void)ws_size;
    const float* x  = (const float*)d_in[0];
    const float* gw = (const float*)d_in[1];
    const float* gb = (const float*)d_in[2];
    const float* w1 = (const float*)d_in[3];
    const float* b1 = (const float*)d_in[4];
    const float* w2 = (const float*)d_in[5];
    const float* b2 = (const float*)d_in[6];
    float* out = (float*)d_out;

    char* ws = (char*)d_ws;
    size_t off = 0;
    __bf16* w1s = (__bf16*)(ws + off); off += (size_t)E * DIN * DHID * 2;   // 64 MB
    __bf16* w2s = (__bf16*)(ws + off); off += (size_t)E * DHID * DOUT * 2;  // 64 MB
    int*   counts = (int*)(ws + off);  off += 256;
    int*   ids    = (int*)(ws + off);  off += (size_t)E * NTOK * sizeof(int);
    float* tws    = (float*)(ws + off);

    moe_zero<<<4096, 256, 0, stream>>>(out, NTOK * DOUT, counts);
    moe_swz<<<8192, 256, 0, stream>>>(w1, w1s, DIN, DHID);
    moe_swz<<<8192, 256, 0, stream>>>(w2, w2s, DHID, DOUT);
    moe_gate<<<NTOK / 8, 256, 0, stream>>>(x, gw, gb, counts, ids, tws);

    size_t shmem = LDS_XFRAG_BYTES + LDS_HFRAG_BYTES + LDS_YACC_BYTES
                 + MT * 16 * (sizeof(int) + sizeof(float));
    dim3 grid(NTOK / (MT * 16), E);
    moe_ffn<<<grid, 512, shmem, stream>>>(x, b1, b2, (const v16bf*)w1s, (const v16bf*)w2s,
                                          counts, ids, tws, out);
}
